// krnn_conv_local1_39204461478647
// MI455X (gfx1250) — matrix operand order repaired and checked by reading, not compiled, not run on hardware
//
#include <hip/hip_runtime.h>

typedef __attribute__((ext_vector_type(16))) __bf16 v16bf;
typedef __attribute__((ext_vector_type(8)))  float  v8f;

#define BN_TOT 51200
#define NBR    9
#define TOUT   12

__device__ __forceinline__ unsigned short f2bf(float f) {
  unsigned int u = __builtin_bit_cast(unsigned int, f);
  u = (u + 0x7FFFu + ((u >> 16) & 1u)) >> 16;   // round-to-nearest-even
  return (unsigned short)u;
}
__device__ __forceinline__ unsigned int packbf2(float lo, float hi) {
  return (unsigned int)f2bf(lo) | ((unsigned int)f2bf(hi) << 16);
}
__device__ __forceinline__ v16bf ld_frag(const unsigned int* p0, const unsigned int* p1) {
  union { v16bf v; uint4 q[2]; } f;
  f.q[0] = *(const uint4*)p0;
  f.q[1] = *(const uint4*)p1;
  return f.v;
}
__device__ __forceinline__ v8f wmma_bf16(v16bf a, v16bf b, v8f c) {
  // D = A(16x32 bf16) * B(32x16 bf16) + C(16x16 f32)
  return __builtin_amdgcn_wmma_f32_16x16x32_bf16(false, a, false, b, (short)0, c, false, false);
}
__device__ __forceinline__ float sigm(float x) {
  return __builtin_amdgcn_rcpf(1.f + __expf(-x));
}
__device__ __forceinline__ float tanh_f(float x) {
  x = fminf(fmaxf(x, -15.f), 15.f);
  float e = __expf(-2.f * x);
  return (1.f - e) * __builtin_amdgcn_rcpf(1.f + e);
}

// Stage a 64x192 (gh = h @ w^T) weight matrix as 12 col-tiles x 2 K-fragments of
// 32x16 bf16 B operands, 8 contiguous dwords per lane, pre-swizzled for ds_load_b128.
__device__ __forceinline__ void stage_whhB(const float* __restrict__ whh,
                                           unsigned int* sB, int tid) {
  for (int idx = tid; idx < 12 * 2 * 32 * 8; idx += 128) {
    int v = idx & 7;
    int l = (idx >> 3) & 31;
    int f = (idx >> 8) & 1;
    int j = idx >> 9;
    int k0  = f * 32 + 2 * v + ((l & 16) ? 8 : 0) + ((v & 4) ? 8 : 0);
    int col = j * 16 + (l & 15);
    sB[idx] = packbf2(whh[col * 64 + k0], whh[col * 64 + k0 + 1]);
  }
}

__global__ __launch_bounds__(128, 1)
void krnn_gru_branch_kernel(
    const float* __restrict__ X,        // (BN, 15, 8)
    const float* __restrict__ convw2, const float* __restrict__ convw3,
    const float* __restrict__ convw4, const float* __restrict__ convb,
    const float* __restrict__ enc_wih, const float* __restrict__ enc_whh,
    const float* __restrict__ enc_bih, const float* __restrict__ enc_bhh,
    const float* __restrict__ dec_wih, const float* __restrict__ dec_whh,
    const float* __restrict__ dec_bih, const float* __restrict__ dec_bhh,
    const float* __restrict__ lin_w,  const float* __restrict__ lin_b,
    float* __restrict__ outBranch)      // (NBR, TOUT, BN)
{
  __shared__ __attribute__((aligned(16))) unsigned int sBwhh[12 * 2 * 32 * 8]; // 24KB enc whh, reused dec whh
  __shared__ __attribute__((aligned(16))) unsigned int sBwih[12 * 32 * 8];     // 12KB enc wih (K padded 8->32)
  __shared__ __attribute__((aligned(16))) unsigned int sHu[4][16 * 32];        // bf16 h, row-major, per wave
  __shared__ __attribute__((aligned(16))) unsigned int sXp[4][16 * 16];        // bf16 x_t padded 16x32, per wave
  __shared__ __attribute__((aligned(16))) float sBE[384];   // enc bih | bhh
  __shared__ __attribute__((aligned(16))) float sBD[384];   // dec bih | bhh
  __shared__ __attribute__((aligned(16))) float sDwih[192];
  __shared__ __attribute__((aligned(16))) float sLw[64];
  __shared__ __attribute__((aligned(16))) float sCW[256];   // conv W (O,I,K)
  __shared__ __attribute__((aligned(16))) float sCB[8];
  __shared__ __attribute__((aligned(16))) float sLv[4][16];
  __shared__ float sLbv;

  const int r    = blockIdx.y;
  const int tid  = threadIdx.x;
  const int w    = tid >> 5;
  const int lane = tid & 31;
  const int rowi = lane & 15;
  const int hi16 = lane >> 4;
  const int seqBase = blockIdx.x * 64 + w * 16;

  const int Ks[NBR] = {2, 2, 2, 3, 3, 3, 4, 4, 4};
  const int Dd[NBR] = {1, 2, 4, 1, 2, 4, 1, 2, 4};
  const int K = Ks[r], D = Dd[r];
  const int L = 15 - D * (K - 1);

  // ---------------- staging ----------------
  stage_whhB(enc_whh + (size_t)r * 192 * 64, sBwhh, tid);
  {
    const float* wih = enc_wih + (size_t)r * 192 * 8;
    for (int idx = tid; idx < 12 * 32 * 8; idx += 128) {
      int v = idx & 7;
      int l = (idx >> 3) & 31;
      int j = idx >> 8;
      int k0  = 2 * v + ((l & 16) ? 8 : 0) + ((v & 4) ? 8 : 0);
      int col = j * 16 + (l & 15);
      float lo = (k0     < 8) ? wih[col * 8 + k0]     : 0.f;
      float hi = (k0 + 1 < 8) ? wih[col * 8 + k0 + 1] : 0.f;
      sBwih[idx] = packbf2(lo, hi);
    }
  }
  for (int i = tid; i < 192; i += 128) {
    sBE[i]       = enc_bih[r * 192 + i];
    sBE[192 + i] = enc_bhh[r * 192 + i];
    sBD[i]       = dec_bih[r * 192 + i];
    sBD[192 + i] = dec_bhh[r * 192 + i];
    sDwih[i]     = dec_wih[r * 192 + i];
  }
  for (int i = tid; i < 64; i += 128) sLw[i] = lin_w[r * 64 + i];
  if (tid == 0) sLbv = lin_b[r];
  {
    const float* cw = (r < 3) ? convw2 + r * 128
                    : (r < 6) ? convw3 + (r - 3) * 192
                              : convw4 + (r - 6) * 256;
    for (int i = tid; i < 64 * K; i += 128) sCW[i] = cw[i];
    for (int i = tid; i < 8; i += 128) sCB[i] = convb[r * 8 + i];
  }
  for (int i = lane; i < 256; i += 32) sXp[w][i] = 0u;  // zero pad (K=8..31 stays zero)
  for (int i = lane; i < 512; i += 32) sHu[w][i] = 0u;  // h0 = 0
  __syncthreads();

  float hreg[4][8];
  #pragma unroll
  for (int cg = 0; cg < 4; ++cg)
    #pragma unroll
    for (int v = 0; v < 8; ++v) hreg[cg][v] = 0.f;

  const unsigned int* hb0 = &sHu[w][rowi * 32 + hi16 * 4];
  unsigned short* hS = (unsigned short*)&sHu[w][0];

  // ---------------- encoder ----------------
  for (int t = 0; t < L; ++t) {
    // dilated conv on the fly: lane computes 4 of 8 channels for its row
    {
      const float* xrow = X + (size_t)(seqBase + rowi) * 120;
      const int fog = hi16 * 4;
      float c0 = sCB[fog + 0], c1 = sCB[fog + 1], c2 = sCB[fog + 2], c3 = sCB[fog + 3];
      for (int k = 0; k < K; ++k) {
        const float* xp = xrow + (t + k * D) * 8;
        #pragma unroll
        for (int fi = 0; fi < 8; ++fi) {
          float xv = xp[fi];
          c0 = fmaf(xv, sCW[((fog + 0) * 8 + fi) * K + k], c0);
          c1 = fmaf(xv, sCW[((fog + 1) * 8 + fi) * K + k], c1);
          c2 = fmaf(xv, sCW[((fog + 2) * 8 + fi) * K + k], c2);
          c3 = fmaf(xv, sCW[((fog + 3) * 8 + fi) * K + k], c3);
        }
      }
      unsigned int* xpw = &sXp[w][rowi * 16 + hi16 * 2];
      xpw[0] = packbf2(c0, c1);
      xpw[1] = packbf2(c2, c3);
    }
    asm volatile("s_wait_dscnt 0" ::: "memory");

    // A fragments (16x32 bf16): x_t and the two K-halves of h
    const unsigned int* xb = &sXp[w][rowi * 16 + hi16 * 4];
    v16bf xA  = ld_frag(xb, xb + 8);
    v16bf hA0 = ld_frag(hb0, hb0 + 8);
    v16bf hA1 = ld_frag(hb0 + 16, hb0 + 24);

    #pragma unroll
    for (int cg = 0; cg < 4; ++cg) {
      const int colH = cg * 16 + rowi;
      const int jR = cg, jZ = 4 + cg, jN = 8 + cg;
      float bR  = sBE[colH]       + sBE[192 + colH];
      float bZ  = sBE[64 + colH]  + sBE[256 + colH];
      float bNX = sBE[128 + colH];
      float bNH = sBE[320 + colH];
      v8f aR, aZ, aNX, aNH;
      #pragma unroll
      for (int v = 0; v < 8; ++v) { aR[v] = bR; aZ[v] = bZ; aNX[v] = bNX; aNH[v] = bNH; }

      const unsigned int* pwR = &sBwih[(jR * 32 + lane) * 8];
      const unsigned int* pwZ = &sBwih[(jZ * 32 + lane) * 8];
      const unsigned int* pwN = &sBwih[(jN * 32 + lane) * 8];
      aR  = wmma_bf16(xA, ld_frag(pwR, pwR + 4), aR);
      aZ  = wmma_bf16(xA, ld_frag(pwZ, pwZ + 4), aZ);
      aNX = wmma_bf16(xA, ld_frag(pwN, pwN + 4), aNX);

      const unsigned int* phR0 = &sBwhh[((jR * 2 + 0) * 32 + lane) * 8];
      const unsigned int* phR1 = &sBwhh[((jR * 2 + 1) * 32 + lane) * 8];
      const unsigned int* phZ0 = &sBwhh[((jZ * 2 + 0) * 32 + lane) * 8];
      const unsigned int* phZ1 = &sBwhh[((jZ * 2 + 1) * 32 + lane) * 8];
      const unsigned int* phN0 = &sBwhh[((jN * 2 + 0) * 32 + lane) * 8];
      const unsigned int* phN1 = &sBwhh[((jN * 2 + 1) * 32 + lane) * 8];
      aR  = wmma_bf16(hA0, ld_frag(phR0, phR0 + 4), aR);
      aR  = wmma_bf16(hA1, ld_frag(phR1, phR1 + 4), aR);
      aZ  = wmma_bf16(hA0, ld_frag(phZ0, phZ0 + 4), aZ);
      aZ  = wmma_bf16(hA1, ld_frag(phZ1, phZ1 + 4), aZ);
      aNH = wmma_bf16(hA0, ld_frag(phN0, phN0 + 4), aNH);
      aNH = wmma_bf16(hA1, ld_frag(phN1, phN1 + 4), aNH);

      #pragma unroll
      for (int v = 0; v < 8; ++v) {
        float rg = sigm(aR[v]);
        float zg = sigm(aZ[v]);
        float ng = tanh_f(aNX[v] + rg * aNH[v]);
        float hv = (1.f - zg) * ng + zg * hreg[cg][v];
        hreg[cg][v] = hv;
        hS[(v + hi16 * 8) * 64 + colH] = f2bf(hv);
      }
    }
    asm volatile("s_wait_dscnt 0" ::: "memory");
  }

  // ---------------- restage decoder whh into the same B buffer ----------------
  __syncthreads();
  stage_whhB(dec_whh + (size_t)r * 192 * 64, sBwhh, tid);
  if (lane < 16) sLv[w][lane] = X[(size_t)(seqBase + lane) * 120 + 112]; // last0 = X[:,14,0]
  __syncthreads();

  // ---------------- decoder ----------------
  for (int td = 0; td < TOUT; ++td) {
    float4 lva = *(const float4*)&sLv[w][hi16 * 8];
    float4 lvb = *(const float4*)&sLv[w][hi16 * 8 + 4];
    float lvv[8] = {lva.x, lva.y, lva.z, lva.w, lvb.x, lvb.y, lvb.z, lvb.w};

    v16bf hA0 = ld_frag(hb0, hb0 + 8);
    v16bf hA1 = ld_frag(hb0 + 16, hb0 + 24);

    float part[8];
    #pragma unroll
    for (int v = 0; v < 8; ++v) part[v] = 0.f;

    #pragma unroll
    for (int cg = 0; cg < 4; ++cg) {
      const int colH = cg * 16 + rowi;
      const int jR = cg, jZ = 4 + cg, jN = 8 + cg;
      float dwR = sDwih[colH], dwZ = sDwih[64 + colH], dwN = sDwih[128 + colH];
      float bR  = sBD[colH]      + sBD[192 + colH];
      float bZ  = sBD[64 + colH] + sBD[256 + colH];
      float bNX = sBD[128 + colH];
      float bNH = sBD[320 + colH];
      v8f aR, aZ, aNH;
      #pragma unroll
      for (int v = 0; v < 8; ++v) {
        aR[v]  = fmaf(lvv[v], dwR, bR);   // rank-1 input GEMM in VALU
        aZ[v]  = fmaf(lvv[v], dwZ, bZ);
        aNH[v] = bNH;
      }
      const unsigned int* phR0 = &sBwhh[((jR * 2 + 0) * 32 + lane) * 8];
      const unsigned int* phR1 = &sBwhh[((jR * 2 + 1) * 32 + lane) * 8];
      const unsigned int* phZ0 = &sBwhh[((jZ * 2 + 0) * 32 + lane) * 8];
      const unsigned int* phZ1 = &sBwhh[((jZ * 2 + 1) * 32 + lane) * 8];
      const unsigned int* phN0 = &sBwhh[((jN * 2 + 0) * 32 + lane) * 8];
      const unsigned int* phN1 = &sBwhh[((jN * 2 + 1) * 32 + lane) * 8];
      aR  = wmma_bf16(hA0, ld_frag(phR0, phR0 + 4), aR);
      aR  = wmma_bf16(hA1, ld_frag(phR1, phR1 + 4), aR);
      aZ  = wmma_bf16(hA0, ld_frag(phZ0, phZ0 + 4), aZ);
      aZ  = wmma_bf16(hA1, ld_frag(phZ1, phZ1 + 4), aZ);
      aNH = wmma_bf16(hA0, ld_frag(phN0, phN0 + 4), aNH);
      aNH = wmma_bf16(hA1, ld_frag(phN1, phN1 + 4), aNH);

      float lwv = sLw[colH];
      #pragma unroll
      for (int v = 0; v < 8; ++v) {
        float rg = sigm(aR[v]);
        float zg = sigm(aZ[v]);
        float nx = fmaf(lvv[v], dwN, bNX);
        float ng = tanh_f(nx + rg * aNH[v]);
        float hv = (1.f - zg) * ng + zg * hreg[cg][v];
        hreg[cg][v] = hv;
        hS[(v + hi16 * 8) * 64 + colH] = f2bf(hv);
        part[v] = fmaf(hv, lwv, part[v]);
      }
    }

    // 64->1 linear: reduce over the 16 lanes of each half-wave
    #pragma unroll
    for (int m = 1; m < 16; m <<= 1) {
      #pragma unroll
      for (int v = 0; v < 8; ++v) part[v] += __shfl_xor(part[v], m, 32);
    }
    float lb = sLbv;
    if ((lane & 15) == 0) {
      float* op = outBranch + ((size_t)(r * TOUT + td)) * BN_TOT + seqBase + hi16 * 8;
      float4 o0 = {part[0] + lb, part[1] + lb, part[2] + lb, part[3] + lb};
      float4 o1 = {part[4] + lb, part[5] + lb, part[6] + lb, part[7] + lb};
      *(float4*)op       = o0;
      *(float4*)(op + 4) = o1;
      *(float4*)&sLv[w][hi16 * 8]     = o0;   // lv feedback
      *(float4*)&sLv[w][hi16 * 8 + 4] = o1;
    }
    asm volatile("s_wait_dscnt 0" ::: "memory");
  }
}

__global__ __launch_bounds__(256)
void krnn_combine_kernel(const float* __restrict__ branch,  // (NBR, TOUT, BN)
                         const float* __restrict__ embed,   // (800, NBR)
                         float* __restrict__ out)           // (BN, TOUT)
{
  int i = blockIdx.x * blockDim.x + threadIdx.x;
  if (i >= BN_TOT * TOUT) return;
  int bn = i / TOUT;
  int t  = i - bn * TOUT;
  int n  = bn % 800;
  float e[NBR], mx = -3.4e38f;
  #pragma unroll
  for (int c = 0; c < NBR; ++c) { e[c] = embed[n * NBR + c]; mx = fmaxf(mx, e[c]); }
  float s = 0.f;
  #pragma unroll
  for (int c = 0; c < NBR; ++c) { e[c] = __expf(e[c] - mx); s += e[c]; }
  float inv = __builtin_amdgcn_rcpf(s);
  float acc = 0.f;
  #pragma unroll
  for (int c = 0; c < NBR; ++c)
    acc = fmaf(e[c] * inv, branch[((size_t)(c * TOUT + t)) * BN_TOT + bn], acc);
  out[i] = acc;
}

extern "C" void kernel_launch(void* const* d_in, const int* in_sizes, int n_in,
                              void* d_out, int out_size, void* d_ws, size_t ws_size,
                              hipStream_t stream) {
  (void)in_sizes; (void)n_in; (void)out_size; (void)ws_size;
  const float* X       = (const float*)d_in[1];
  const float* convw2  = (const float*)d_in[2];
  const float* convw3  = (const float*)d_in[3];
  const float* convw4  = (const float*)d_in[4];
  const float* convb   = (const float*)d_in[5];
  const float* enc_wih = (const float*)d_in[6];
  const float* enc_whh = (const float*)d_in[7];
  const float* enc_bih = (const float*)d_in[8];
  const float* enc_bhh = (const float*)d_in[9];
  const float* dec_wih = (const float*)d_in[10];
  const float* dec_whh = (const float*)d_in[11];
  const float* dec_bih = (const float*)d_in[12];
  const float* dec_bhh = (const float*)d_in[13];
  const float* lin_w   = (const float*)d_in[14];
  const float* lin_b   = (const float*)d_in[15];
  const float* embed   = (const float*)d_in[16];

  float* branch = (float*)d_ws;   // NBR * TOUT * BN floats = 22.1 MB

  dim3 grid(BN_TOT / 64, NBR);
  krnn_gru_branch_kernel<<<grid, 128, 0, stream>>>(
      X, convw2, convw3, convw4, convb,
      enc_wih, enc_whh, enc_bih, enc_bhh,
      dec_wih, dec_whh, dec_bih, dec_bhh,
      lin_w, lin_b, branch);

  int total = BN_TOT * TOUT;
  krnn_combine_kernel<<<(total + 255) / 256, 256, 0, stream>>>(branch, embed, (float*)d_out);
}
